// GIN_28123445854590
// MI455X (gfx1250) — matrix operand order repaired and checked by reading, not compiled, not run on hardware
//
#include <hip/hip_runtime.h>
#include <hip/hip_bf16.h>

typedef __attribute__((ext_vector_type(16))) _Float16 v16h;
typedef __attribute__((ext_vector_type(8)))  _Float16 v8h;
typedef __attribute__((ext_vector_type(4)))  _Float16 v4h;
typedef __attribute__((ext_vector_type(8)))  float    v8f;

// 32-bit LDS (address_space(3)) pointer to an 8 x f16 vector
typedef __attribute__((address_space(3))) const v8h* lds_v8h_ptr;

#define D_FEAT 128
#define N_CLS  10
#define BN_EPS 1e-5f
#define SW_STRIDE 136   // LDS row stride in f16 (128 + 8 pad to break 256B bank conflicts)

// ---------------------------------------------------------------------------
// Utility: zero an f32 buffer
__global__ void gin_zero_f32(float* p, long n) {
    long i = (long)blockIdx.x * blockDim.x + threadIdx.x;
    if (i < n) p[i] = 0.0f;
}

// ---------------------------------------------------------------------------
// Convert + transpose weights: Wt[l][n][k] = W[l][k][n], f32 -> f16
__global__ void gin_weight_tr(const float* __restrict__ W, _Float16* __restrict__ Wt, int total) {
    int i = blockIdx.x * blockDim.x + threadIdx.x;
    if (i >= total) return;
    int l = i / (D_FEAT * D_FEAT);
    int r = i - l * (D_FEAT * D_FEAT);
    int n = r / D_FEAT;
    int k = r - n * D_FEAT;
    Wt[i] = (_Float16)W[(long)l * D_FEAT * D_FEAT + (long)k * D_FEAT + n];
}

// ---------------------------------------------------------------------------
// Convert f32 x -> f16 (4 elements / thread)
__global__ void gin_cvt_f16(const float* __restrict__ x, _Float16* __restrict__ o, long n4) {
    long i = (long)blockIdx.x * blockDim.x + threadIdx.x;
    if (i >= n4) return;
    float4 v = *(const float4*)(x + i * 4);
    v4h h; h.x = (_Float16)v.x; h.y = (_Float16)v.y; h.z = (_Float16)v.z; h.w = (_Float16)v.w;
    *(v4h*)(o + i * 4) = h;
}

// ---------------------------------------------------------------------------
// z = h  (f16 copy, 8 f16 = 16B per thread)
__global__ void gin_copy16(const _Float16* __restrict__ h, _Float16* __restrict__ z, long n16) {
    long i = (long)blockIdx.x * blockDim.x + threadIdx.x;
    if (i >= n16) return;
    *((uint4*)z + i) = *((const uint4*)h + i);
}

// ---------------------------------------------------------------------------
// Edge scatter: z[dst] += h[src]  via packed f16 atomics (2 f16 per op).
// One thread handles one edge's 4-feature chunk (2 pk atomics).
__global__ void gin_scatter(const _Float16* __restrict__ h, const int* __restrict__ ei,
                            _Float16* __restrict__ z, long E) {
    long gid = (long)blockIdx.x * blockDim.x + threadIdx.x;
    long e = gid >> 5;
    if (e >= E) return;
    int c4 = (int)(gid & 31) * 4;
    int src = ei[e];
    int dst = ei[E + e];
    uint2 d = *(const uint2*)(h + (long)src * D_FEAT + c4);
    _Float16* zp = z + (long)dst * D_FEAT + c4;
    asm volatile("global_atomic_pk_add_f16 %0, %1, off" :: "v"(zp),     "v"(d.x) : "memory");
    asm volatile("global_atomic_pk_add_f16 %0, %1, off" :: "v"(zp + 2), "v"(d.y) : "memory");
}

// ---------------------------------------------------------------------------
// WMMA GEMM: Y[N x 128] = A[N x 128](f16) @ W(f16, pre-transposed [n][k]) + bias, f32 out.
// Block = 256 threads = 8 wave32. Block slab: 64 rows x 128 cols.
// W (32KB) is staged into LDS once per block via gfx1250 async copy
// (global_load_async_to_lds_b128, ASYNCcnt). B fragments are read back through
// a laundered address_space(3) pointer so they lower to ds_load_b128 and the
// optimizer cannot fold them (the async writes are invisible to alias analysis).
__global__ __launch_bounds__(256)
void gin_gemm_wmma(const _Float16* __restrict__ A, const _Float16* __restrict__ Wt,
                   const float* __restrict__ bias, float* __restrict__ Y, int nrows) {
    __shared__ _Float16 sW[D_FEAT * SW_STRIDE];

    // ---- async stage: 128x128 f16, 16B chunks. 2048 chunks / 256 threads = 8 iters.
    {
        const int t = threadIdx.x;
        #pragma unroll
        for (int j = 0; j < 8; ++j) {
            int idx = j * 256 + t;          // chunk id (8 f16 per chunk)
            int n   = idx >> 4;             // row (0..127)
            int cw  = idx & 15;             // 16B chunk within row
            _Float16* lp = sW + n * SW_STRIDE + cw * 8;
            // low 32 bits of the generic address == wave-relative LDS byte address
            unsigned lds_addr = (unsigned)(unsigned long long)lp;
            const _Float16* gp = Wt + (long)n * D_FEAT + cw * 8;
            asm volatile("global_load_async_to_lds_b128 %0, %1, off"
                         :: "v"(lds_addr), "v"(gp) : "memory");
        }
        asm volatile("s_wait_asynccnt 0x0" ::: "memory");
    }
    __syncthreads();

    // Laundered 32-bit LDS base offset: provenance-opaque, loads via AS(3)
    // lower to ds_load_b128 and can never be constant-folded.
    unsigned sbase = (unsigned)(unsigned long long)(_Float16*)sW;
    asm volatile("" : "+v"(sbase));

    const int wave = threadIdx.x >> 5;
    const int lane = threadIdx.x & 31;
    const int half = lane >> 4;     // half-wave select
    const int ln   = lane & 15;     // M (for A/C) or N (for B/C)
    const int rowBase  = blockIdx.x * 64 + (wave & 3) * 16;
    const int colStart = (wave >> 2) * 64;

    int arow = rowBase + ln;
    if (arow > nrows - 1) arow = nrows - 1;   // clamp, keep EXEC full for WMMA

    v8f acc0 = {}, acc1 = {}, acc2 = {}, acc3 = {};

    #pragma unroll
    for (int kc = 0; kc < D_FEAT; kc += 32) {
        // A fragment (16x32 f16): elems 0..7 = K kc+half*8.., elems 8..15 = K kc+16+half*8..
        const _Float16* ap = A + (long)arow * D_FEAT + kc + half * 8;
        v8h alo = *(const v8h*)ap;
        v8h ahi = *(const v8h*)(ap + 16);
        v16h af = __builtin_shufflevector(alo, ahi, 0,1,2,3,4,5,6,7,8,9,10,11,12,13,14,15);

        const int bk = kc + half * 8;    // elems 0..7 = K kc+half*8.., elems 8..15 = K kc+16+half*8..
        #pragma unroll
        for (int t = 0; t < 4; ++t) {
            unsigned boff = sbase + (unsigned)(((colStart + t * 16 + ln) * SW_STRIDE + bk) * 2);
            lds_v8h_ptr bp = (lds_v8h_ptr)boff;
            v8h blo = bp[0];
            v8h bhi = bp[2];
            v16h bf = __builtin_shufflevector(blo, bhi, 0,1,2,3,4,5,6,7,8,9,10,11,12,13,14,15);
            v8f c = (t == 0) ? acc0 : (t == 1) ? acc1 : (t == 2) ? acc2 : acc3;
            c = __builtin_amdgcn_wmma_f32_16x16x32_f16(false, af, false, bf,
                                                       (short)0, c, false, false);
            if (t == 0) acc0 = c; else if (t == 1) acc1 = c; else if (t == 2) acc2 = c; else acc3 = c;
        }
    }

    // Store: VGPR r holds M = r + 8*half, N = ln (per 16x16 f32 C layout)
    #pragma unroll
    for (int t = 0; t < 4; ++t) {
        const v8f c = (t == 0) ? acc0 : (t == 1) ? acc1 : (t == 2) ? acc2 : acc3;
        const int col = colStart + t * 16 + ln;
        const float bb = bias[col];
        #pragma unroll
        for (int r = 0; r < 8; ++r) {
            int row = rowBase + half * 8 + r;
            if (row < nrows) Y[(long)row * D_FEAT + col] = c[r] + bb;
        }
    }
}

// ---------------------------------------------------------------------------
// Column sum / sumsq over Y[N x 128] -> stats[0..127]=sum, [128..255]=sumsq
__global__ void gin_colstats(const float* __restrict__ Y, float* __restrict__ stats, int nrows) {
    int c = threadIdx.x & 127;
    int r0 = blockIdx.x * 2 + (threadIdx.x >> 7);
    int stride = gridDim.x * 2;
    float s = 0.f, sq = 0.f;
    for (int r = r0; r < nrows; r += stride) {
        float v = Y[(long)r * D_FEAT + c];
        s += v; sq += v * v;
    }
    atomicAdd(&stats[c], s);
    atomicAdd(&stats[D_FEAT + c], sq);
}

// stats[256..383]=mean, [384..511]=rstd
__global__ void gin_finstats(float* stats, int nrows) {
    int c = threadIdx.x;
    if (c >= D_FEAT) return;
    float inv = 1.0f / (float)nrows;
    float m = stats[c] * inv;
    float v = stats[D_FEAT + c] * inv - m * m;
    stats[2 * D_FEAT + c] = m;
    stats[3 * D_FEAT + c] = rsqrtf(v + BN_EPS);
}

// out = relu((Y - mean)*rstd*g + b) as f16, 4 elems / thread
__global__ void gin_bn_relu(const float* __restrict__ Y, const float* __restrict__ stats,
                            const float* __restrict__ g, const float* __restrict__ b,
                            _Float16* __restrict__ out, long n4) {
    long i = (long)blockIdx.x * blockDim.x + threadIdx.x;
    if (i >= n4) return;
    int c = (int)((i * 4) & (D_FEAT - 1));
    float4 y = *(const float4*)(Y + i * 4);
    const float* mean = stats + 2 * D_FEAT;
    const float* rstd = stats + 3 * D_FEAT;
    float r0 = (y.x - mean[c + 0]) * rstd[c + 0] * g[c + 0] + b[c + 0];
    float r1 = (y.y - mean[c + 1]) * rstd[c + 1] * g[c + 1] + b[c + 1];
    float r2 = (y.z - mean[c + 2]) * rstd[c + 2] * g[c + 2] + b[c + 2];
    float r3 = (y.w - mean[c + 3]) * rstd[c + 3] * g[c + 3] + b[c + 3];
    v4h h;
    h.x = (_Float16)fmaxf(r0, 0.f); h.y = (_Float16)fmaxf(r1, 0.f);
    h.z = (_Float16)fmaxf(r2, 0.f); h.w = (_Float16)fmaxf(r3, 0.f);
    *(v4h*)(out + i * 4) = h;
}

// ---------------------------------------------------------------------------
// Graph pooling: pooled[buf][batch[n]][c] += outs[buf][n][c]  (f32 atomics)
__global__ void gin_pool(const _Float16* __restrict__ outs, const int* __restrict__ batch,
                         float* __restrict__ pooled, int N, int G) {
    int buf = blockIdx.y;
    long gid = (long)blockIdx.x * blockDim.x + threadIdx.x;
    long node = gid >> 5;
    if (node >= N) return;
    int c4 = (int)(gid & 31) * 4;
    int gph = batch[node];
    const _Float16* p = outs + ((long)buf * N + node) * D_FEAT + c4;
    float* q = pooled + ((long)buf * G + gph) * D_FEAT + c4;
    atomicAdd(q + 0, (float)p[0]);
    atomicAdd(q + 1, (float)p[1]);
    atomicAdd(q + 2, (float)p[2]);
    atomicAdd(q + 3, (float)p[3]);
}

// ---------------------------------------------------------------------------
// Per-graph heads + log_softmax: one thread per graph
__global__ void gin_heads(const float* __restrict__ pooled, const float* __restrict__ fcW,
                          const float* __restrict__ fcb, float* __restrict__ out,
                          int G, int NH) {
    int gph = blockIdx.x * blockDim.x + threadIdx.x;
    if (gph >= G) return;
    float logit[N_CLS];
    #pragma unroll
    for (int c = 0; c < N_CLS; ++c) logit[c] = 0.f;
    for (int i = 0; i < NH; ++i) {
        #pragma unroll
        for (int c = 0; c < N_CLS; ++c) logit[c] += fcb[i * N_CLS + c];
        const float* pv = pooled + ((long)i * G + gph) * D_FEAT;
        const float* wv = fcW + (long)i * D_FEAT * N_CLS;
        for (int d = 0; d < D_FEAT; ++d) {
            float p = pv[d];
            #pragma unroll
            for (int c = 0; c < N_CLS; ++c) logit[c] += p * wv[d * N_CLS + c];
        }
    }
    float mx = logit[0];
    #pragma unroll
    for (int c = 1; c < N_CLS; ++c) mx = fmaxf(mx, logit[c]);
    float se = 0.f;
    #pragma unroll
    for (int c = 0; c < N_CLS; ++c) se += __expf(logit[c] - mx);
    float lse = __logf(se);
    #pragma unroll
    for (int c = 0; c < N_CLS; ++c) out[(long)gph * N_CLS + c] = logit[c] - mx - lse;
}

// ---------------------------------------------------------------------------
extern "C" void kernel_launch(void* const* d_in, const int* in_sizes, int n_in,
                              void* d_out, int out_size, void* d_ws, size_t ws_size,
                              hipStream_t stream) {
    const float* x        = (const float*)d_in[0];
    const int*   ei       = (const int*)d_in[1];
    const int*   batch    = (const int*)d_in[2];
    const float* conv_W1  = (const float*)d_in[4];
    const float* conv_b1  = (const float*)d_in[5];
    const float* cbn_g    = (const float*)d_in[6];
    const float* cbn_b    = (const float*)d_in[7];
    const float* conv_W2  = (const float*)d_in[8];
    const float* conv_b2  = (const float*)d_in[9];
    const float* bn_g     = (const float*)d_in[10];
    const float* bn_b     = (const float*)d_in[11];
    const float* fc_W     = (const float*)d_in[12];
    const float* fc_b     = (const float*)d_in[13];

    const int  N = in_sizes[0] / D_FEAT;
    const long E = (long)in_sizes[1] / 2;
    const int  L = in_sizes[4] / (D_FEAT * D_FEAT);
    const int  G = out_size / N_CLS;
    const int  NH = L + 1;

    // Workspace carve-up (256B aligned)
    size_t off = 0;
    auto carve = [&](size_t bytes) -> void* {
        off = (off + 255) & ~(size_t)255;
        void* p = (char*)d_ws + off;
        off += bytes;
        return p;
    };
    _Float16* W1t    = (_Float16*)carve((size_t)L * D_FEAT * D_FEAT * 2);
    _Float16* W2t    = (_Float16*)carve((size_t)L * D_FEAT * D_FEAT * 2);
    _Float16* outs   = (_Float16*)carve((size_t)NH * N * D_FEAT * 2);
    _Float16* zbuf   = (_Float16*)carve((size_t)N * D_FEAT * 2);
    _Float16* abuf   = (_Float16*)carve((size_t)N * D_FEAT * 2);
    float*    ybuf   = (float*)   carve((size_t)N * D_FEAT * 4);
    float*    stats  = (float*)   carve(4 * D_FEAT * 4);
    float*    pooled = (float*)   carve((size_t)NH * G * D_FEAT * 4);
    (void)ws_size;

    const int wTot = L * D_FEAT * D_FEAT;
    gin_weight_tr<<<(wTot + 255) / 256, 256, 0, stream>>>(conv_W1, W1t, wTot);
    gin_weight_tr<<<(wTot + 255) / 256, 256, 0, stream>>>(conv_W2, W2t, wTot);

    const long n4  = (long)N * D_FEAT / 4;
    const long n16 = (long)N * D_FEAT / 8;
    gin_cvt_f16<<<(int)((n4 + 255) / 256), 256, 0, stream>>>(x, outs, n4);

    const int  gemmBlocks   = (N + 63) / 64;
    const long scatThreads  = E * 32;
    const int  scatBlocks   = (int)((scatThreads + 255) / 256);

    for (int l = 0; l < L; ++l) {
        _Float16* h = outs + (long)l * N * D_FEAT;
        // z = h + scatter_add(h[src] -> dst)
        gin_copy16<<<(int)((n16 + 255) / 256), 256, 0, stream>>>(h, zbuf, n16);
        gin_scatter<<<scatBlocks, 256, 0, stream>>>(h, ei, zbuf, E);
        // GEMM1 + BN + relu
        gin_gemm_wmma<<<gemmBlocks, 256, 0, stream>>>(zbuf, W1t + (long)l * D_FEAT * D_FEAT,
                                                      conv_b1 + l * D_FEAT, ybuf, N);
        gin_zero_f32<<<1, 256, 0, stream>>>(stats, 2 * D_FEAT);
        gin_colstats<<<256, 256, 0, stream>>>(ybuf, stats, N);
        gin_finstats<<<1, 128, 0, stream>>>(stats, N);
        gin_bn_relu<<<(int)((n4 + 255) / 256), 256, 0, stream>>>(
            ybuf, stats, cbn_g + l * D_FEAT, cbn_b + l * D_FEAT, abuf, n4);
        // GEMM2 + outer BN + relu -> next h
        gin_gemm_wmma<<<gemmBlocks, 256, 0, stream>>>(abuf, W2t + (long)l * D_FEAT * D_FEAT,
                                                      conv_b2 + l * D_FEAT, ybuf, N);
        gin_zero_f32<<<1, 256, 0, stream>>>(stats, 2 * D_FEAT);
        gin_colstats<<<256, 256, 0, stream>>>(ybuf, stats, N);
        gin_finstats<<<1, 128, 0, stream>>>(stats, N);
        gin_bn_relu<<<(int)((n4 + 255) / 256), 256, 0, stream>>>(
            ybuf, stats, bn_g + l * D_FEAT, bn_b + l * D_FEAT,
            outs + (long)(l + 1) * N * D_FEAT, n4);
    }

    // Pooling over 5 scales + heads + log_softmax
    const long pTot = (long)NH * G * D_FEAT;
    gin_zero_f32<<<(int)((pTot + 255) / 256), 256, 0, stream>>>(pooled, pTot);
    dim3 pg((unsigned)(((long)N * 32 + 255) / 256), (unsigned)NH, 1);
    gin_pool<<<pg, 256, 0, stream>>>(outs, batch, pooled, N, G);
    gin_heads<<<(G + 255) / 256, 256, 0, stream>>>(pooled, fc_W, fc_b, (float*)d_out, G, NH);
}
